// A2aSparseStackedMlp_35983236006084
// MI455X (gfx1250) — matrix operand order repaired and checked by reading, not compiled, not run on hardware
//
#include <hip/hip_runtime.h>
#include <hip/hip_bf16.h>

// ---------------------------------------------------------------------------
// MoE (gpt-oss style) forward for MI455X / gfx1250, wave32 + WMMA bf16.
//   output [B,S,H] then router_scores [B,S,E] concatenated in d_out (fp32).
// Sparse token-grouped expert compute: only the K=4 selected experts per
// token are evaluated (math identical to the masked-dense reference).
// GEMMs run on v_wmma_f32_16x16x32_bf16 with fp32 accumulation.
//  - A operands (bf16) staged to LDS with global_load_async_to_lds_b128
//    (ASYNCcnt path, drained with s_wait_asynccnt before the barrier).
//  - B operands (fp32 weights) converted to bf16 in-register while staging,
//    packed as (k,k+1) pairs -> b32 DS stores; next tile prefetched.
// ---------------------------------------------------------------------------

typedef __bf16 bf16_t;
typedef __bf16 v16bf __attribute__((ext_vector_type(16)));
typedef __bf16 v8bf  __attribute__((ext_vector_type(8)));
typedef __bf16 v2bf  __attribute__((ext_vector_type(2)));
typedef float  v8f   __attribute__((ext_vector_type(8)));

constexpr int B_ = 2, S_ = 512, H_ = 1024, I_ = 1024, E_ = 16, K_ = 4;
constexpr int T_ = B_ * S_;          // 1024 tokens
constexpr int MT = 64;               // pair-rows per M tile
constexpr int CAP = 5120;            // padded pair capacity >= 4096 + 16*(MT-1)
constexpr int MAXTILES = CAP / MT;   // 80
constexpr float ALPHA = 1.702f;
constexpr float LIMIT = 7.0f;

// ---------------- workspace layout (bytes, 256-aligned sections) -----------
constexpr size_t OFF_HIDBF   = 0;                                   // T*H bf16
constexpr size_t OFF_ACT     = OFF_HIDBF + (size_t)T_ * H_ * 2;     // CAP*I bf16
constexpr size_t OFF_DOUT    = OFF_ACT + (size_t)CAP * I_ * 2;      // CAP*H f32
constexpr size_t OFF_TOPKI   = OFF_DOUT + (size_t)CAP * H_ * 4;     // T*K int
constexpr size_t OFF_TOPKW   = OFF_TOPKI + (size_t)T_ * K_ * 4;     // T*K f32
constexpr size_t OFF_POSOF   = OFF_TOPKW + (size_t)T_ * K_ * 4;     // T*K int
constexpr size_t OFF_PAIRTOK = OFF_POSOF + (size_t)T_ * K_ * 4;     // CAP int
constexpr size_t OFF_COUNTS  = OFF_PAIRTOK + (size_t)CAP * 4;       // E int
constexpr size_t OFF_CURSOR  = OFF_COUNTS + 256;                    // E int
constexpr size_t OFF_POFF    = OFF_CURSOR + 256;                    // E+1 int
constexpr size_t OFF_TILEEXP = OFF_POFF + 256;                      // MAXTILES int

// Async copy of 16 bytes global -> LDS, tracked on ASYNCcnt.
__device__ __forceinline__ void async_copy_b128(unsigned ldsAddr,
                                                unsigned long long gAddr) {
    asm volatile("global_load_async_to_lds_b128 %0, %1, off"
                 :: "v"(ldsAddr), "v"(gAddr) : "memory");
}
__device__ __forceinline__ void wait_async0() {
    asm volatile("s_wait_asynccnt 0x0" ::: "memory");
}
__device__ __forceinline__ unsigned lds_addr_of(const void* p) {
    return (unsigned)(uintptr_t)p;   // low 32 bits of flat shared addr = LDS offset
}

// ---------------------------------------------------------------------------
__global__ __launch_bounds__(256)
void init_kernel(int* __restrict__ pairTok, int* __restrict__ counts) {
    int g = blockIdx.x * 256 + threadIdx.x;
    if (g < CAP) pairTok[g] = -1;
    if (g < E_) counts[g] = 0;
}

// ---------------------------------------------------------------------------
__global__ __launch_bounds__(256)
void router_kernel(const float* __restrict__ hid, const float* __restrict__ rw,
                   const float* __restrict__ rb, float* __restrict__ scoresOut,
                   int* __restrict__ topkIdx, float* __restrict__ topkW,
                   int* __restrict__ counts) {
    int t = blockIdx.x * 256 + threadIdx.x;
    if (t >= T_) return;

    float acc[E_];
#pragma unroll
    for (int e = 0; e < E_; ++e) acc[e] = rb[e];

    const float* hrow = hid + (size_t)t * H_;
    for (int h = 0; h < H_; ++h) {
        float x = hrow[h];
        const float4* w = (const float4*)(rw + (size_t)h * E_);
        float4 w0 = w[0], w1 = w[1], w2 = w[2], w3 = w[3];
        acc[0]  += x * w0.x; acc[1]  += x * w0.y; acc[2]  += x * w0.z; acc[3]  += x * w0.w;
        acc[4]  += x * w1.x; acc[5]  += x * w1.y; acc[6]  += x * w1.z; acc[7]  += x * w1.w;
        acc[8]  += x * w2.x; acc[9]  += x * w2.y; acc[10] += x * w2.z; acc[11] += x * w2.w;
        acc[12] += x * w3.x; acc[13] += x * w3.y; acc[14] += x * w3.z; acc[15] += x * w3.w;
    }

    // top-4 (first-occurrence ties like jax.lax.top_k), softmax over selected
    int idx[K_]; float val[K_];
    unsigned taken = 0;
#pragma unroll
    for (int k = 0; k < K_; ++k) {
        int bi = 0; float bv = -3.0e38f;
#pragma unroll
        for (int e = 0; e < E_; ++e)
            if (!((taken >> e) & 1u) && acc[e] > bv) { bv = acc[e]; bi = e; }
        taken |= 1u << bi; idx[k] = bi; val[k] = bv;
    }
    float m = val[0], s = 0.0f, w[K_];
#pragma unroll
    for (int k = 0; k < K_; ++k) { w[k] = __expf(val[k] - m); s += w[k]; }
    float inv = 1.0f / s;
#pragma unroll
    for (int k = 0; k < K_; ++k) w[k] *= inv;

    float sc[E_];
#pragma unroll
    for (int e = 0; e < E_; ++e) sc[e] = 0.0f;
#pragma unroll
    for (int k = 0; k < K_; ++k) sc[idx[k]] = w[k];
#pragma unroll
    for (int e = 0; e < E_; ++e) scoresOut[(size_t)t * E_ + e] = sc[e];
#pragma unroll
    for (int k = 0; k < K_; ++k) {
        topkIdx[t * K_ + k] = idx[k];
        topkW[t * K_ + k] = w[k];
        atomicAdd(&counts[idx[k]], 1);
    }
}

// ---------------------------------------------------------------------------
__global__ void scan_kernel(const int* __restrict__ counts, int* __restrict__ poff,
                            int* __restrict__ cursor, int* __restrict__ tileExp) {
    if (threadIdx.x == 0) {
        int off = 0;
        for (int e = 0; e < E_; ++e) {
            poff[e] = off;
            cursor[e] = off;
            off += ((counts[e] + MT - 1) / MT) * MT;
        }
        poff[E_] = off;
        for (int tile = 0; tile < MAXTILES; ++tile) {
            int r = tile * MT, te = -1;
            for (int q = 0; q < E_; ++q)
                if (r >= poff[q] && r < poff[q + 1]) { te = q; break; }
            tileExp[tile] = te;
        }
    }
}

// ---------------------------------------------------------------------------
__global__ __launch_bounds__(256)
void assign_kernel(const int* __restrict__ topkIdx, int* __restrict__ cursor,
                   int* __restrict__ pairTok, int* __restrict__ posOf) {
    int t = blockIdx.x * 256 + threadIdx.x;
    if (t >= T_) return;
#pragma unroll
    for (int k = 0; k < K_; ++k) {
        int e = topkIdx[t * K_ + k];
        int pos = atomicAdd(&cursor[e], 1);
        pairTok[pos] = t;
        posOf[t * K_ + k] = pos;
    }
}

// ---------------------------------------------------------------------------
__global__ __launch_bounds__(256)
void cvt_hidden_kernel(const float* __restrict__ hid, bf16_t* __restrict__ out) {
    int i = blockIdx.x * 256 + threadIdx.x;   // grid covers T*H exactly
    out[i] = (bf16_t)hid[i];
}

// ---------------------------------------------------------------------------
// gate_up GEMM + activation.  Tile: 64 pair-rows x 64 act cols per WG.
// 4 waves; wave w owns N-subtile w (16 cols) for BOTH gate and up halves.
__global__ __launch_bounds__(128)
void gateup_kernel(const bf16_t* __restrict__ hid,   // [T][H] bf16
                   const float* __restrict__ gup,    // [E][H][2I]
                   const float* __restrict__ gub,    // [E][2I]
                   const int* __restrict__ pairTok,  // [CAP]
                   const int* __restrict__ tileExp,  // [MAXTILES]
                   bf16_t* __restrict__ act)         // [CAP][I]
{
    const int mtile = blockIdx.x;
    const int e = tileExp[mtile];
    if (e < 0) return;
    const int n0 = blockIdx.y * 64;      // act column base
    const int rowBase = mtile * MT;

    __shared__ bf16_t aS[MT][32];        // A K-slice (rows x 32 K)
    __shared__ bf16_t bS[128][32];       // col-major: [0..63]=gate, [64..127]=up
    __shared__ int tokS[MT];

    const int tid = threadIdx.x;
    const int lane = tid & 31;
    const int wave = tid >> 5;

    if (tid < MT) {
        int tk = pairTok[rowBase + tid];
        tokS[tid] = (tk < 0) ? 0 : tk;
    }
    __syncthreads();

    v8f cg[4] = {v8f{}, v8f{}, v8f{}, v8f{}};
    v8f cu[4] = {v8f{}, v8f{}, v8f{}, v8f{}};

    const float* gupE = gup + (size_t)e * H_ * (2 * I_);

    for (int k0 = 0; k0 < H_; k0 += 32) {
        // stage A: 64 rows x 32 K of bf16 -> LDS via async copy (ASYNCcnt)
        for (int c = tid; c < MT * 4; c += 128) {
            int r = c >> 2, q = c & 3;
            async_copy_b128(lds_addr_of(&aS[r][q * 8]),
                            (unsigned long long)(uintptr_t)(
                                hid + (size_t)tokS[r] * H_ + k0 + q * 8));
        }
        // stage B: rows (k0..k0+31) x (64 gate + 64 up) fp32 -> bf16 col-major.
        // Each thread handles a K-row pair so (k,k+1) pack into one b32 store.
        for (int c = tid; c < 512; c += 128) {
            int kp = c >> 5;           // 0..15  (K-row pair)
            int hv = (c >> 4) & 1;     // 0=gate, 1=up
            int c4 = c & 15;           // float4 index in 64 cols
            int gcol = (hv ? (I_ + n0) : n0) + c4 * 4;
            const float* base = gupE + (size_t)(k0 + 2 * kp) * (2 * I_) + gcol;
            float4 wa = *(const float4*)base;
            float4 wb = *(const float4*)(base + 2 * I_);
            if (k0 + 32 < H_) __builtin_prefetch(base + (size_t)32 * (2 * I_), 0, 0);
            int cl = hv * 64 + c4 * 4;
            v2bf p;
            p[0] = (bf16_t)wa.x; p[1] = (bf16_t)wb.x; *(v2bf*)&bS[cl + 0][2 * kp] = p;
            p[0] = (bf16_t)wa.y; p[1] = (bf16_t)wb.y; *(v2bf*)&bS[cl + 1][2 * kp] = p;
            p[0] = (bf16_t)wa.z; p[1] = (bf16_t)wb.z; *(v2bf*)&bS[cl + 2][2 * kp] = p;
            p[0] = (bf16_t)wa.w; p[1] = (bf16_t)wb.w; *(v2bf*)&bS[cl + 3][2 * kp] = p;
        }
        wait_async0();
        __syncthreads();

        // B fragments: lane -> col = lane%16, K = (lane/16)*16 + 0..15 contiguous
        const int colLocal = wave * 16 + (lane & 15);
        const int kbase = (lane >> 4) * 16;
        v16bf bg = *(const v16bf*)&bS[colLocal][kbase];
        v16bf bu = *(const v16bf*)&bS[64 + colLocal][kbase];

#pragma unroll
        for (int mi = 0; mi < 4; ++mi) {
            // A fragment: lane -> row = mi*16 + lane%16;
            // K elems 0..15 = k0+khalf+0..15 contiguous (khalf = (lane/16)*16)
            const int ar = mi * 16 + (lane & 15);
            const int khalf = (lane >> 4) * 16;
            v8bf alo = *(const v8bf*)&aS[ar][khalf];
            v8bf ahi = *(const v8bf*)&aS[ar][khalf + 8];
            v16bf a;
#pragma unroll
            for (int i = 0; i < 8; ++i) { a[i] = alo[i]; a[8 + i] = ahi[i]; }
            cg[mi] = __builtin_amdgcn_wmma_f32_16x16x32_bf16(
                false, a, false, bg, (short)0, cg[mi], false, false);
            cu[mi] = __builtin_amdgcn_wmma_f32_16x16x32_bf16(
                false, a, false, bu, (short)0, cu[mi], false, false);
        }
        __syncthreads();
    }

    // epilogue: bias + gpt-oss activation, write act bf16
    const int col = n0 + wave * 16 + (lane & 15);
    const float gb = gub[(size_t)e * 2 * I_ + col];
    const float ub = gub[(size_t)e * 2 * I_ + I_ + col];
#pragma unroll
    for (int mi = 0; mi < 4; ++mi) {
#pragma unroll
        for (int v = 0; v < 8; ++v) {
            float g = cg[mi][v] + gb;
            float u = cu[mi][v] + ub;
            g = fminf(g, LIMIT);
            u = fminf(fmaxf(u, -LIMIT), LIMIT);
            float sig = 1.0f / (1.0f + __expf(-ALPHA * g));
            float a = (u + 1.0f) * (g * sig);
            int row = rowBase + mi * 16 + ((lane < 16) ? v : v + 8);
            act[(size_t)row * I_ + col] = (bf16_t)a;
        }
    }
}

// ---------------------------------------------------------------------------
// down GEMM: act[pos][I] (bf16) x down_proj[e][I][H] + down_bias -> dout fp32
__global__ __launch_bounds__(128)
void down_kernel(const bf16_t* __restrict__ act,   // [CAP][I]
                 const float* __restrict__ dwn,    // [E][I][H]
                 const float* __restrict__ dwb,    // [E][H]
                 const int* __restrict__ tileExp,
                 float* __restrict__ dout)         // [CAP][H]
{
    const int mtile = blockIdx.x;
    const int e = tileExp[mtile];
    if (e < 0) return;
    const int n0 = blockIdx.y * 64;
    const int rowBase = mtile * MT;

    __shared__ bf16_t aS[MT][32];
    __shared__ bf16_t bS[64][32];

    const int tid = threadIdx.x;
    const int lane = tid & 31;
    const int wave = tid >> 5;

    v8f cc[4] = {v8f{}, v8f{}, v8f{}, v8f{}};
    const float* dwnE = dwn + (size_t)e * I_ * H_;

    for (int k0 = 0; k0 < I_; k0 += 32) {
        // stage A (act bf16) via async copy
        for (int c = tid; c < MT * 4; c += 128) {
            int r = c >> 2, q = c & 3;
            async_copy_b128(lds_addr_of(&aS[r][q * 8]),
                            (unsigned long long)(uintptr_t)(
                                act + (size_t)(rowBase + r) * I_ + k0 + q * 8));
        }
        // stage B: K-row pairs, packed b32 stores
        for (int c = tid; c < 256; c += 128) {
            int kp = c >> 4;      // 0..15
            int c4 = c & 15;      // float4 index in 64 cols
            const float* base = dwnE + (size_t)(k0 + 2 * kp) * H_ + n0 + c4 * 4;
            float4 wa = *(const float4*)base;
            float4 wb = *(const float4*)(base + H_);
            if (k0 + 32 < I_) __builtin_prefetch(base + (size_t)32 * H_, 0, 0);
            int cl = c4 * 4;
            v2bf p;
            p[0] = (bf16_t)wa.x; p[1] = (bf16_t)wb.x; *(v2bf*)&bS[cl + 0][2 * kp] = p;
            p[0] = (bf16_t)wa.y; p[1] = (bf16_t)wb.y; *(v2bf*)&bS[cl + 1][2 * kp] = p;
            p[0] = (bf16_t)wa.z; p[1] = (bf16_t)wb.z; *(v2bf*)&bS[cl + 2][2 * kp] = p;
            p[0] = (bf16_t)wa.w; p[1] = (bf16_t)wb.w; *(v2bf*)&bS[cl + 3][2 * kp] = p;
        }
        wait_async0();
        __syncthreads();

        const int colLocal = wave * 16 + (lane & 15);
        const int kbase = (lane >> 4) * 16;
        v16bf b = *(const v16bf*)&bS[colLocal][kbase];

#pragma unroll
        for (int mi = 0; mi < 4; ++mi) {
            const int ar = mi * 16 + (lane & 15);
            const int khalf = (lane >> 4) * 16;
            v8bf alo = *(const v8bf*)&aS[ar][khalf];
            v8bf ahi = *(const v8bf*)&aS[ar][khalf + 8];
            v16bf a;
#pragma unroll
            for (int i = 0; i < 8; ++i) { a[i] = alo[i]; a[8 + i] = ahi[i]; }
            cc[mi] = __builtin_amdgcn_wmma_f32_16x16x32_bf16(
                false, a, false, b, (short)0, cc[mi], false, false);
        }
        __syncthreads();
    }

    const int col = n0 + wave * 16 + (lane & 15);
    const float bias = dwb[(size_t)e * H_ + col];
#pragma unroll
    for (int mi = 0; mi < 4; ++mi) {
#pragma unroll
        for (int v = 0; v < 8; ++v) {
            int row = rowBase + mi * 16 + ((lane < 16) ? v : v + 8);
            dout[(size_t)row * H_ + col] = cc[mi][v] + bias;
        }
    }
}

// ---------------------------------------------------------------------------
__global__ __launch_bounds__(128)
void combine_kernel(const float* __restrict__ dout, const int* __restrict__ posOf,
                    const float* __restrict__ topkW, float* __restrict__ out) {
    int t = blockIdx.x;
    int p0 = posOf[t * K_ + 0], p1 = posOf[t * K_ + 1];
    int p2 = posOf[t * K_ + 2], p3 = posOf[t * K_ + 3];
    float w0 = topkW[t * K_ + 0], w1 = topkW[t * K_ + 1];
    float w2 = topkW[t * K_ + 2], w3 = topkW[t * K_ + 3];
    for (int h = threadIdx.x; h < H_; h += 128) {
        float a = w0 * dout[(size_t)p0 * H_ + h] + w1 * dout[(size_t)p1 * H_ + h] +
                  w2 * dout[(size_t)p2 * H_ + h] + w3 * dout[(size_t)p3 * H_ + h];
        out[(size_t)t * H_ + h] = a;
    }
}

// ---------------------------------------------------------------------------
extern "C" void kernel_launch(void* const* d_in, const int* in_sizes, int n_in,
                              void* d_out, int out_size, void* d_ws, size_t ws_size,
                              hipStream_t stream) {
    const float* hid = (const float*)d_in[0];
    const float* rw  = (const float*)d_in[1];
    const float* rb  = (const float*)d_in[2];
    const float* gup = (const float*)d_in[3];
    const float* gub = (const float*)d_in[4];
    const float* dwn = (const float*)d_in[5];
    const float* dwb = (const float*)d_in[6];

    char* ws = (char*)d_ws;
    bf16_t* hidBf  = (bf16_t*)(ws + OFF_HIDBF);
    bf16_t* actBf  = (bf16_t*)(ws + OFF_ACT);
    float*  doutWs = (float*)(ws + OFF_DOUT);
    int*    topkI  = (int*)(ws + OFF_TOPKI);
    float*  topkW  = (float*)(ws + OFF_TOPKW);
    int*    posOf  = (int*)(ws + OFF_POSOF);
    int*    pairTok= (int*)(ws + OFF_PAIRTOK);
    int*    counts = (int*)(ws + OFF_COUNTS);
    int*    cursor = (int*)(ws + OFF_CURSOR);
    int*    poff   = (int*)(ws + OFF_POFF);
    int*    tileEx = (int*)(ws + OFF_TILEEXP);

    float* out = (float*)d_out;
    float* scores = out + (size_t)T_ * H_;

    init_kernel<<<(CAP + 255) / 256, 256, 0, stream>>>(pairTok, counts);
    cvt_hidden_kernel<<<(T_ * H_) / 256, 256, 0, stream>>>(hid, hidBf);
    router_kernel<<<T_ / 256, 256, 0, stream>>>(hid, rw, rb, scores, topkI, topkW, counts);
    scan_kernel<<<1, 64, 0, stream>>>(counts, poff, cursor, tileEx);
    assign_kernel<<<T_ / 256, 256, 0, stream>>>(topkI, cursor, pairTok, posOf);
    gateup_kernel<<<dim3(MAXTILES, I_ / 64), 128, 0, stream>>>(
        hidBf, gup, gub, pairTok, tileEx, actBf);
    down_kernel<<<dim3(MAXTILES, H_ / 64), 128, 0, stream>>>(
        actBf, dwn, dwb, tileEx, doutWs);
    combine_kernel<<<T_, 128, 0, stream>>>(doutWs, posOf, topkW, out);
}
